// MoChA_68719476736381
// MI455X (gfx1250) — matrix operand order repaired and checked by reading, not compiled, not run on hardware
//
#include <hip/hip_runtime.h>
#include <hip/hip_bf16.h>
#include <stdint.h>

// ---------------- problem constants ----------------
static constexpr int B_  = 16;
static constexpr int T_  = 4096;
static constexpr int KD_ = 1024;
static constexpr int QD_ = 1024;
static constexpr int AD_ = 512;
static constexpr int VD_ = 1024;
static constexpr int W_  = 8;
static constexpr int NTOT = 2 * AD_;   // 1024 combined columns (mono | chunk)

// ---------------- WMMA types ----------------
typedef __attribute__((ext_vector_type(16))) __bf16 v16bf;
typedef __attribute__((ext_vector_type(8)))  float  v8f;

union V16U { uint32_t u[8]; uint4 q[2]; v16bf v; };

// round-to-nearest-even f32 -> bf16 bits (manual, avoids relying on __bf16 casts)
__device__ __forceinline__ uint32_t bf16_rne(float x) {
  uint32_t u = __builtin_bit_cast(uint32_t, x);
  return (u + 0x7FFFu + ((u >> 16) & 1u)) >> 16;
}
__device__ __forceinline__ float bf16_to_f32(uint32_t h) {
  return __builtin_bit_cast(float, h << 16);
}
__device__ __forceinline__ uint32_t pack_hi(float a, float b) {
  return bf16_rne(a) | (bf16_rne(b) << 16);
}
__device__ __forceinline__ uint32_t pack_lo(float a, float b) {
  float ra = a - bf16_to_f32(bf16_rne(a));
  float rb = b - bf16_to_f32(bf16_rne(b));
  return bf16_rne(ra) | (bf16_rne(rb) << 16);
}

// ---------------- prep: split combined [KD,1024] weights into bf16 hi/lo, WMMA-fragment order ----
// W2 layout: [KD/32][1024][32] of uint32.  For k-block kb, column n:
//   inner[0..15]  = hi-packed k-pairs (kpi = kp & 15, kp = global k/2)
//   inner[16..31] = lo-packed k-pairs
// A lane's B fragment (8 dwords, given its K-half) is contiguous & 32B aligned.
__global__ __launch_bounds__(256) void prep_weights(
    const float* __restrict__ Wk_mono, const float* __restrict__ Wk_chunk,
    uint32_t* __restrict__ W2) {
  int idx = blockIdx.x * 256 + threadIdx.x;          // over (KD/2)*1024 k-pairs x cols
  if (idx >= (KD_ / 2) * NTOT) return;
  int kp = idx >> 10;                                // global k-pair 0..511
  int n  = idx & (NTOT - 1);
  int k0 = 2 * kp, k1 = 2 * kp + 1;
  float w0, w1;
  if (n < AD_) { w0 = Wk_mono[k0 * AD_ + n];          w1 = Wk_mono[k1 * AD_ + n]; }
  else         { w0 = Wk_chunk[k0 * AD_ + (n - AD_)]; w1 = Wk_chunk[k1 * AD_ + (n - AD_)]; }
  int kb  = kp >> 4;                                 // 32-k block
  int kpi = kp & 15;                                 // k-pair within block
  size_t base = ((size_t)kb * NTOT + n) * 32;
  W2[base + kpi]      = pack_hi(w0, w1);
  W2[base + 16 + kpi] = pack_lo(w0, w1);
}

// ---------------- prep: qmc[b][n] = bias(n) + query[b]·Wq(:,n)  (n<512: mono, else chunk) --------
__global__ __launch_bounds__(256) void qmc_kernel(
    const float* __restrict__ query, const float* __restrict__ Wq_mono,
    const float* __restrict__ Wq_chunk, const float* __restrict__ b_mono,
    const float* __restrict__ b_chunk, float* __restrict__ qmc) {
  int n = blockIdx.x * 256 + threadIdx.x;            // 0..1023
  int b = blockIdx.y;
  const float* q = query + (size_t)b * QD_;
  float s;
  if (n < AD_) {
    s = b_mono[n];
    for (int k = 0; k < QD_; ++k) s += q[k] * Wq_mono[k * AD_ + n];
  } else {
    int nn = n - AD_;
    s = b_chunk[nn];
    for (int k = 0; k < QD_; ++k) s += q[k] * Wq_chunk[k * AD_ + nn];
  }
  qmc[(size_t)b * NTOT + n] = s;
}

// ---------------- stage 1: fused GEMM (bf16x3) + tanh·v epilogue -> e_mono, e_chunk --------------
// block = 256 threads (8 waves); 32 rows of key per block; wave w owns N columns [128w,128w+128)
// Each B fragment (4 x b128 loads) feeds 6 WMMAs (2 M-tiles x bf16x3).
__global__ __launch_bounds__(256) void fused_energy(
    const float* __restrict__ key,
    const uint32_t* __restrict__ W2,
    const float* __restrict__ qmc,
    const float* __restrict__ v_mono, const float* __restrict__ v_chunk,
    const float* __restrict__ r_mono, const float* __restrict__ r_chunk,
    float* __restrict__ e_mono, float* __restrict__ e_chunk) {
  __shared__ uint32_t Ah[2][32 * 16];   // double-buffered: 32 rows x 16 k-pairs (hi)
  __shared__ uint32_t Al[2][32 * 16];   // (lo)
  __shared__ float    part[8][32];      // per-wave row partials

  const int tid  = threadIdx.x;
  const int wave = tid >> 5;
  const int lane = tid & 31;
  const int half = lane >> 4;
  const int ln   = lane & 15;
  const int mrow = blockIdx.x * 32;          // global row base (b*T + t)
  const int batch = mrow / T_;
  const int nbase = wave * 128;

  v8f zero = {};
  v8f acc0[8], acc1[8];
#pragma unroll
  for (int j = 0; j < 8; ++j) { acc0[j] = zero; acc1[j] = zero; }

  // loader mapping: thread -> (row lm, float4 kq)
  const int lm = tid >> 3;                   // 0..31
  const int kq = tid & 7;                    // 0..7 (float4 per row)
  const float* keyrow = key + (size_t)(mrow + lm) * KD_;

  // lane's A fragment base offsets in the staged tile (k-pair runs of 4, 16B aligned)
  const int arun0 = 8 * half;                // kp run {8h..8h+3}
  const int arun1 = 8 * half + 4;            // kp run {8h+4..8h+7}

  int pb = 0;
  for (int kstep = 0; kstep < KD_; kstep += 32) {
    // ---- load 32x32 f32 A tile, split to bf16 hi/lo, stage in LDS ----
    float4 kv = *(const float4*)(keyrow + kstep + 4 * kq);
    uint32_t hi0 = pack_hi(kv.x, kv.y), lo0 = pack_lo(kv.x, kv.y);
    uint32_t hi1 = pack_hi(kv.z, kv.w), lo1 = pack_lo(kv.z, kv.w);
    Ah[pb][lm * 16 + 2 * kq]     = hi0;
    Ah[pb][lm * 16 + 2 * kq + 1] = hi1;
    Al[pb][lm * 16 + 2 * kq]     = lo0;
    Al[pb][lm * 16 + 2 * kq + 1] = lo1;
    __syncthreads();

    // ---- read A fragments (two 16-row tiles) ----
    V16U a0h, a0l, a1h, a1l;
    {
      const uint32_t* p0 = &Ah[pb][ln * 16];
      const uint32_t* p1 = &Ah[pb][(16 + ln) * 16];
      const uint32_t* q0 = &Al[pb][ln * 16];
      const uint32_t* q1 = &Al[pb][(16 + ln) * 16];
      a0h.q[0] = *(const uint4*)(p0 + arun0);  a0h.q[1] = *(const uint4*)(p0 + arun1);
      a1h.q[0] = *(const uint4*)(p1 + arun0);  a1h.q[1] = *(const uint4*)(p1 + arun1);
      a0l.q[0] = *(const uint4*)(q0 + arun0);  a0l.q[1] = *(const uint4*)(q0 + arun1);
      a1l.q[0] = *(const uint4*)(q1 + arun0);  a1l.q[1] = *(const uint4*)(q1 + arun1);
    }

    const uint32_t* Wblk = W2 + (size_t)(kstep >> 5) * NTOT * 32;
#pragma unroll
    for (int j = 0; j < 8; ++j) {
      int n = nbase + 16 * j + ln;
      const uint32_t* bp = Wblk + (size_t)n * 32 + half * 8;  // 32B-aligned runs
      V16U b_hi, b_lo;
      b_hi.q[0] = *(const uint4*)(bp);
      b_hi.q[1] = *(const uint4*)(bp + 4);
      b_lo.q[0] = *(const uint4*)(bp + 16);
      b_lo.q[1] = *(const uint4*)(bp + 20);
      acc0[j] = __builtin_amdgcn_wmma_f32_16x16x32_bf16(false, a0h.v, false, b_hi.v,
                                                        (short)0, acc0[j], false, false);
      acc0[j] = __builtin_amdgcn_wmma_f32_16x16x32_bf16(false, a0h.v, false, b_lo.v,
                                                        (short)0, acc0[j], false, false);
      acc0[j] = __builtin_amdgcn_wmma_f32_16x16x32_bf16(false, a0l.v, false, b_hi.v,
                                                        (short)0, acc0[j], false, false);
      acc1[j] = __builtin_amdgcn_wmma_f32_16x16x32_bf16(false, a1h.v, false, b_hi.v,
                                                        (short)0, acc1[j], false, false);
      acc1[j] = __builtin_amdgcn_wmma_f32_16x16x32_bf16(false, a1h.v, false, b_lo.v,
                                                        (short)0, acc1[j], false, false);
      acc1[j] = __builtin_amdgcn_wmma_f32_16x16x32_bf16(false, a1l.v, false, b_hi.v,
                                                        (short)0, acc1[j], false, false);
    }
    pb ^= 1;
  }

  // ---- epilogue: tanh(acc + qmc) · v, reduce over this wave's 128 columns ----
  float rs0[8], rs1[8];
#pragma unroll
  for (int r = 0; r < 8; ++r) { rs0[r] = 0.f; rs1[r] = 0.f; }
#pragma unroll
  for (int j = 0; j < 8; ++j) {
    int n = nbase + 16 * j + ln;
    float qv = qmc[(size_t)batch * NTOT + n];
    float vv = (n < AD_) ? v_mono[n] : v_chunk[n - AD_];
#pragma unroll
    for (int r = 0; r < 8; ++r) {
      rs0[r] += tanhf(acc0[j][r] + qv) * vv;
      rs1[r] += tanhf(acc1[j][r] + qv) * vv;
    }
  }
  // reduce across the 16 lanes of each half (xor 1,2,4,8 stay within a half)
#pragma unroll
  for (int off = 1; off < 16; off <<= 1) {
#pragma unroll
    for (int r = 0; r < 8; ++r) {
      rs0[r] += __shfl_xor(rs0[r], off, 32);
      rs1[r] += __shfl_xor(rs1[r], off, 32);
    }
  }
  if (ln == 0) {
#pragma unroll
    for (int r = 0; r < 8; ++r) {
      part[wave][8 * half + r]      = rs0[r];   // rows 0..15
      part[wave][16 + 8 * half + r] = rs1[r];   // rows 16..31
    }
  }
  __syncthreads();
  if (tid < 64) {
    int row   = tid & 31;
    int which = tid >> 5;                    // 0 = mono (waves 0-3), 1 = chunk (waves 4-7)
    float s = 0.f;
#pragma unroll
    for (int w = 0; w < 4; ++w) s += part[which * 4 + w][row];
    if (which == 0) e_mono[mrow + row] = s + r_mono[0];
    else            e_chunk[mrow + row] = s + r_chunk[0];
  }
}

// ---------------- stage 2: per-batch scan (cumsum / max / window sums) -> aw, beta --------------
__global__ __launch_bounds__(256) void scan_kernel(
    const float* __restrict__ e_mono, const float* __restrict__ e_chunk,
    const float* __restrict__ noise, float* __restrict__ aw_out,
    float* __restrict__ beta) {
  __shared__ float s_scan[256];
  __shared__ float s_red[256];
  __shared__ float sexp_sh[T_];
  __shared__ float ratio_sh[T_];
  const int b = blockIdx.x, tid = threadIdx.x;
  const float* em = e_mono  + (size_t)b * T_;
  const float* ec = e_chunk + (size_t)b * T_;
  const float* nz = noise   + (size_t)b * T_;
  const int t0 = tid * 16;

  // p_choose + local prefix of log(clip(1-p))
  float p[16], pre[16];
  float s = 0.f;
#pragma unroll
  for (int i = 0; i < 16; ++i) {
    float e  = em[t0 + i] + nz[t0 + i];
    float pc = 1.f / (1.f + expf(-e));
    p[i]   = pc;
    pre[i] = s;
    float l = logf(fminf(fmaxf(1.f - pc, 1e-10f), 1.f));
    s += l;
  }
  s_scan[tid] = s;
  __syncthreads();
  for (int off = 1; off < 256; off <<= 1) {
    float v = (tid >= off) ? s_scan[tid - off] : 0.f;
    __syncthreads();
    s_scan[tid] += v;
    __syncthreads();
  }
  float base = s_scan[tid] - s;              // exclusive prefix over thread chunks

  // aw[t] = p[t] * exp(1 + sum_{i<t} log1mp[i])   (replicates reference's clip/ones quirk)
  float aw_loc[16];
#pragma unroll
  for (int i = 0; i < 16; ++i) {
    float S = base + pre[i];
    aw_loc[i] = p[i] * expf(1.0f + S);
    aw_out[(size_t)b * T_ + t0 + i] = aw_loc[i];
  }

  // chunk softmax-exp with global max
  float ecl[16];
  float mx = -3.4e38f;
#pragma unroll
  for (int i = 0; i < 16; ++i) { ecl[i] = ec[t0 + i]; mx = fmaxf(mx, ecl[i]); }
  s_red[tid] = mx;
  __syncthreads();
  for (int off = 128; off > 0; off >>= 1) {
    if (tid < off) s_red[tid] = fmaxf(s_red[tid], s_red[tid + off]);
    __syncthreads();
  }
  mx = s_red[0];
  float sexp_loc[16];
#pragma unroll
  for (int i = 0; i < 16; ++i) {
    float v = fmaxf(expf(ecl[i] - mx), 1e-5f);
    sexp_loc[i] = v;
    sexp_sh[t0 + i] = v;
  }
  __syncthreads();
  // ratio[t] = aw[t] / movsum_back7(sexp)[t]
#pragma unroll
  for (int i = 0; i < 16; ++i) {
    int t = t0 + i;
    float d = 0.f;
#pragma unroll
    for (int k = 0; k < W_; ++k) { int tt = t - k; if (tt >= 0) d += sexp_sh[tt]; }
    ratio_sh[t] = aw_loc[i] / d;
  }
  __syncthreads();
  // beta[t] = sexp[t] * movsum_fwd7(ratio)[t]
#pragma unroll
  for (int i = 0; i < 16; ++i) {
    int t = t0 + i;
    float sum = 0.f;
#pragma unroll
    for (int k = 0; k < W_; ++k) { int tt = t + k; if (tt < T_) sum += ratio_sh[tt]; }
    beta[(size_t)b * T_ + t] = sexp_loc[i] * sum;
  }
}

// ---------------- stage 3: cv[b,d] = sum_t beta[b,t] * value[b,t,d] ----------------
__global__ __launch_bounds__(256) void zero_cv(float* __restrict__ cv) {
  int idx = blockIdx.x * 256 + threadIdx.x;
  if (idx < B_ * VD_) cv[idx] = 0.f;
}
__global__ __launch_bounds__(256) void cv_kernel(
    const float* __restrict__ beta, const float* __restrict__ value,
    float* __restrict__ cv) {
  const int d  = blockIdx.x * 256 + threadIdx.x;   // 0..1023 (4 blocks of 256)
  const int tc = blockIdx.y;                       // T chunk (8 of 512)
  const int b  = blockIdx.z;
  const float* vp = value + ((size_t)b * T_ + (size_t)tc * 512) * VD_ + d;
  const float* bp = beta + (size_t)b * T_ + tc * 512;
  float acc = 0.f;
  for (int t = 0; t < 512; ++t) acc += bp[t] * vp[(size_t)t * VD_];
  atomicAdd(&cv[(size_t)b * VD_ + d], acc);
}

// ---------------- launcher ----------------
extern "C" void kernel_launch(void* const* d_in, const int* in_sizes, int n_in,
                              void* d_out, int out_size, void* d_ws, size_t ws_size,
                              hipStream_t stream) {
  const float* key      = (const float*)d_in[0];
  const float* value    = (const float*)d_in[1];
  const float* query    = (const float*)d_in[2];
  const float* noise    = (const float*)d_in[3];
  const float* Wk_mono  = (const float*)d_in[4];
  const float* b_mono   = (const float*)d_in[5];
  const float* Wq_mono  = (const float*)d_in[6];
  const float* Wk_chunk = (const float*)d_in[7];
  const float* b_chunk  = (const float*)d_in[8];
  const float* Wq_chunk = (const float*)d_in[9];
  const float* r_mono   = (const float*)d_in[10];
  const float* r_chunk  = (const float*)d_in[11];
  const float* v_mono   = (const float*)d_in[12];
  const float* v_chunk  = (const float*)d_in[13];

  float* out_cv = (float*)d_out;                       // [B, VD]
  float* out_aw = (float*)d_out + (size_t)B_ * VD_;    // [B, T]

  // workspace carve-up
  uint32_t* W2  = (uint32_t*)d_ws;                     // [KD/32][1024][32] = 4 MB
  float* qmc    = (float*)(W2 + (size_t)(KD_ / 32) * NTOT * 32);   // [B][1024]
  float* e_mono = qmc + (size_t)B_ * NTOT;             // [B*T]
  float* e_chun = e_mono + (size_t)B_ * T_;            // [B*T]
  float* beta   = e_chun + (size_t)B_ * T_;            // [B*T]

  prep_weights<<<(KD_ / 2) * NTOT / 256, 256, 0, stream>>>(Wk_mono, Wk_chunk, W2);
  qmc_kernel<<<dim3(NTOT / 256, B_), 256, 0, stream>>>(query, Wq_mono, Wq_chunk,
                                                       b_mono, b_chunk, qmc);
  fused_energy<<<(B_ * T_) / 32, 256, 0, stream>>>(key, W2, qmc, v_mono, v_chunk,
                                                   r_mono, r_chunk, e_mono, e_chun);
  scan_kernel<<<B_, 256, 0, stream>>>(e_mono, e_chun, noise, out_aw, beta);
  zero_cv<<<(B_ * VD_) / 256, 256, 0, stream>>>(out_cv);
  cv_kernel<<<dim3(VD_ / 256, T_ / 512, B_), 256, 0, stream>>>(beta, value, out_cv);
}
